// MultiHeadedAttention_23742579212869
// MI455X (gfx1250) — compile-verified, matrix operand order checked by reading, not run on hardware
//
#include <hip/hip_runtime.h>

// ---------------------------------------------------------------------------
// MHA (RoPE + rel-pos bias, causal) for MI455X / gfx1250, wave32 WMMA bf16.
// B=2, S=2048, D=1024, H=16, HD=64
// GEMMs: LDS-staged B via global_load_async_to_lds_b128 (ASYNCcnt) + WMMA,
// k-loop unrolled 2x over the double buffer (no register rotation copies).
// ---------------------------------------------------------------------------

typedef __attribute__((ext_vector_type(16))) __bf16 v16bf;
typedef __attribute__((ext_vector_type(8)))  __bf16 v8bf;
typedef __attribute__((ext_vector_type(8)))  float  v8f;

#define BB 2
#define SS 2048
#define DDIM 1024
#define HH 16
#define HD 64
#define MROWS (BB * SS)   // 4096

// ---- CDNA5 async copy helpers (inline asm; portable across toolchains) -----
__device__ __forceinline__ void async_load_b128(unsigned lds_off, const void* gaddr) {
  // VDST = LDS byte address VGPR, VADDR = 64-bit global address, saddr = off
  asm volatile("global_load_async_to_lds_b128 %0, %1, off"
               :: "v"(lds_off), "v"(gaddr) : "memory");
}
__device__ __forceinline__ void wait_asynccnt0() {
  asm volatile("s_wait_asynccnt 0" ::: "memory");
}

// ---- WMMA fragment loaders (layouts per CDNA5 ISA 7.12.2, wave32) ----------
// A 16x32 bf16: lane L: row = L&15, kb = (L>>4)*8; elems 0..7 = K[kb..kb+7],
// elems 8..15 = K[16+kb..16+kb+7].  B 32x16 from a transposed (NxK) source
// has the identical per-lane pattern (row index = output column).
__device__ __forceinline__ v16bf load_frag(const __bf16* base, int stride) {
  const int lane = threadIdx.x & 31;
  const int r = lane & 15;
  const int kb = (lane >> 4) << 3;
  const __bf16* p = base + (size_t)r * stride + kb;
  v8bf lo = *(const v8bf*)(p);
  v8bf hi = *(const v8bf*)(p + 16);
  v16bf f;
#pragma unroll
  for (int i = 0; i < 8; ++i) { f[i] = lo[i]; f[i + 8] = hi[i]; }
  return f;
}

__device__ __forceinline__ v8f wmma_bf16(v16bf a, v16bf b, v8f c) {
  return __builtin_amdgcn_wmma_f32_16x16x32_bf16(
      false, a, false, b, (short)0, c, false, false);
}

// ---- Precision / layout prep ----------------------------------------------
__global__ __launch_bounds__(256) void cvt_bf16_kernel(
    const float* __restrict__ src, __bf16* __restrict__ dst, int n) {
  int i = blockIdx.x * 256 + threadIdx.x;
  if (i < n) dst[i] = (__bf16)src[i];
}

// W[K][N] fp32 -> WT[N][K] bf16  (1024x1024)
__global__ __launch_bounds__(256) void transpose_w_kernel(
    const float* __restrict__ W, __bf16* __restrict__ WT) {
  int i = blockIdx.x * 256 + threadIdx.x;   // 0 .. 2^20-1
  int k = i >> 10, n = i & (DDIM - 1);
  WT[(size_t)n * DDIM + k] = (__bf16)W[i];
}

// ---- WMMA GEMM: C[4096x1024] = A(bf16, row-major) x BT(bf16, NxK) ----------
// Block = 128 threads (4 waves): 4 M-tiles x one shared 64-wide N strip.
// B k-slices staged to LDS with async loads (double buffered, 2x unrolled).
// mode 0: RoPE epilogue -> bf16 [B*H][S][HD]       (Q, K)
// mode 1: transpose epilogue -> bf16 [B*H][HD][S]  (V)
// mode 2: fp32 row-major store [4096][1024]        (final output)
__global__ __launch_bounds__(128) void gemm_bf16_wmma_kernel(
    const __bf16* __restrict__ A, const __bf16* __restrict__ BT,
    void* __restrict__ dst, int mode) {
  __shared__ __align__(16) __bf16 bsh[2][64 * 32];   // 2 x 4KB double buffer

  const int tid  = threadIdx.x;
  const int wid  = tid >> 5;
  const int lane = tid & 31;
  const int hlf  = lane >> 4;
  const int nl   = lane & 15;
  const int mbase = (blockIdx.x >> 4) * 64 + wid * 16;  // 64 M-blocks
  const int nbase = (blockIdx.x & 15) * 64;             // 16 N strips (head-aligned)

  const __bf16* Bt = BT + (size_t)nbase * DDIM;
  const __bf16* Ab = A + (size_t)mbase * DDIM;
  // 64 rows x 64B per k-slice = 256 x 16B chunks; 2 chunks per thread
  const int c0 = tid * 2, c1 = tid * 2 + 1;

  auto stage = [&](int buf, int kk) {
    unsigned lbase = (unsigned)(size_t)(void*)&bsh[buf][0];
    int r0 = c0 >> 2, b0 = (c0 & 3) * 16;
    int r1 = c1 >> 2, b1 = (c1 & 3) * 16;
    async_load_b128(lbase + (unsigned)(r0 * 64 + b0),
                    (const char*)(Bt + (size_t)r0 * DDIM + kk) + b0);
    async_load_b128(lbase + (unsigned)(r1 * 64 + b1),
                    (const char*)(Bt + (size_t)r1 * DDIM + kk) + b1);
  };

  stage(0, 0);
  v16bf a0 = load_frag(Ab, DDIM);
  v16bf a1;
  v8f acc[4] = {};

#pragma unroll 1
  for (int kk = 0; kk < DDIM; kk += 64) {
    // ---- phase 0: consume bsh[0] (k = kk), refill bsh[1] (k = kk+32) ----
    wait_asynccnt0();
    __syncthreads();
    stage(1, kk + 32);                        // kk+32 < 1024 always
    a1 = load_frag(Ab + kk + 32, DDIM);
    {
      const __bf16* bl = &bsh[0][0];
      v16bf b0 = load_frag(bl,        32);    // all 8 ds_load_b128 up front
      v16bf b1 = load_frag(bl +  512, 32);
      v16bf b2 = load_frag(bl + 1024, 32);
      v16bf b3 = load_frag(bl + 1536, 32);
      acc[0] = wmma_bf16(a0, b0, acc[0]);
      acc[1] = wmma_bf16(a0, b1, acc[1]);
      acc[2] = wmma_bf16(a0, b2, acc[2]);
      acc[3] = wmma_bf16(a0, b3, acc[3]);
    }
    // ---- phase 1: consume bsh[1] (k = kk+32), refill bsh[0] (k = kk+64) ----
    wait_asynccnt0();
    __syncthreads();
    const bool more = (kk + 64 < DDIM);
    if (more) {
      stage(0, kk + 64);
      a0 = load_frag(Ab + kk + 64, DDIM);
    }
    {
      const __bf16* bl = &bsh[1][0];
      v16bf b0 = load_frag(bl,        32);
      v16bf b1 = load_frag(bl +  512, 32);
      v16bf b2 = load_frag(bl + 1024, 32);
      v16bf b3 = load_frag(bl + 1536, 32);
      acc[0] = wmma_bf16(a1, b0, acc[0]);
      acc[1] = wmma_bf16(a1, b1, acc[1]);
      acc[2] = wmma_bf16(a1, b2, acc[2]);
      acc[3] = wmma_bf16(a1, b3, acc[3]);
    }
  }

  if (mode == 2) {                                 // fp32 store
    float* O = (float*)dst;
#pragma unroll
    for (int i = 0; i < 8; ++i) {
      int r = mbase + hlf * 8 + i;                 // C-layout: M = i + 8*half
#pragma unroll
      for (int t = 0; t < 4; ++t)
        O[(size_t)r * DDIM + nbase + t * 16 + nl] = acc[t][i];
    }
  } else if (mode == 1) {                          // V -> [BH][HD][S]
    __bf16* O = (__bf16*)dst;
    int h = nbase >> 6;
#pragma unroll
    for (int i = 0; i < 8; ++i) {
      int r = mbase + hlf * 8 + i;
      int b = r >> 11, s = r & (SS - 1);
#pragma unroll
      for (int t = 0; t < 4; ++t) {
        int d = t * 16 + nl;
        O[((size_t)(b * HH + h) * HD + d) * SS + s] = (__bf16)acc[t][i];
      }
    }
  } else {                                         // RoPE -> [BH][S][HD]
    __bf16* O = (__bf16*)dst;
    int h = nbase >> 6;
#pragma unroll
    for (int i = 0; i < 8; ++i) {
      int r = mbase + hlf * 8 + i;
      int b = r >> 11, s = r & (SS - 1);
      __bf16* Orow = O + ((size_t)(b * HH + h) * SS + s) * HD;
#pragma unroll
      for (int tp = 0; tp < 2; ++tp) {
        int d0 = tp * 16 + nl;                     // 0..31; pair is d0+32
        float v0 = acc[tp][i], v1 = acc[tp + 2][i];
        float ang = (float)s * __expf(-(float)d0 * 0.28782313662425572f);
        float cv, sv;
        __sincosf(ang, &sv, &cv);
        Orow[d0]      = (__bf16)(v0 * cv - v1 * sv);
        Orow[d0 + 32] = (__bf16)(v1 * cv + v0 * sv);
      }
    }
  }
}

// ---- Flash attention: per-wave 16-row q tile, 32 keys / iter ---------------
// K fragments software-pipelined one block ahead; V fragment loads issued
// before the softmax VALU work so global latency hides under exp/shuffles.
__global__ __launch_bounds__(128) void attn_wmma_kernel(
    const __bf16* __restrict__ Q,   // [BH][S][HD] (RoPE'd)
    const __bf16* __restrict__ K,   // [BH][S][HD] (RoPE'd)
    const __bf16* __restrict__ Vt,  // [BH][HD][S]
    const float* __restrict__ rel_emb,   // [32][H]
    __bf16* __restrict__ ctx) {          // [B*S][D] bf16
  __shared__ float lut[128];
  __shared__ __align__(16) __bf16 pst[4][16 * 32];  // per-wave P staging

  const int bh = blockIdx.x;           // 0..31
  const int h = bh & (HH - 1);
  const int b = bh >> 4;
  const int tid = threadIdx.x;

  { // relative-position bias LUT (bucket saturates at 31 for dist >= 113)
    int d = tid;
    int bkt;
    if (d < 16) bkt = d;
    else {
      bkt = 16 + (int)(__logf((float)d * (1.0f / 16.0f)) *
                       (16.0f / __logf(8.0f)));
      if (bkt > 31) bkt = 31;
    }
    lut[d] = rel_emb[bkt * HH + h];
  }
  __syncthreads();

  const int wid = tid >> 5, lane = tid & 31;
  const int hlf = lane >> 4, nl = lane & 15;
  const int qbase = blockIdx.y * 64 + wid * 16;

  const __bf16* Qb = Q + ((size_t)bh * SS + qbase) * HD;
  const __bf16* Kb = K + (size_t)bh * SS * HD;
  const __bf16* Vb = Vt + (size_t)bh * HD * SS;

  v16bf aq0 = load_frag(Qb, HD);        // features 0..31
  v16bf aq1 = load_frag(Qb + 32, HD);   // features 32..63

  float m[8], l[8];
  v8f acc[4] = {};
#pragma unroll
  for (int i = 0; i < 8; ++i) { m[i] = -3.0e38f; l[i] = 0.0f; }
  __bf16* myp = &pst[wid][0];

  const int qmax = qbase + 15;
  v16bf bk0 = load_frag(Kb, HD);
  v16bf bk1 = load_frag(Kb + 32, HD);
  v16bf bk2 = load_frag(Kb + 16 * HD, HD);
  v16bf bk3 = load_frag(Kb + 16 * HD + 32, HD);

#pragma unroll 1
  for (int j = 0; j <= qmax; j += 32) {       // keys j..j+31 (always in-range)
    v8f s0 = {}, s1 = {};
    s0 = wmma_bf16(aq0, bk0, s0);
    s0 = wmma_bf16(aq1, bk1, s0);
    s1 = wmma_bf16(aq0, bk2, s1);
    s1 = wmma_bf16(aq1, bk3, s1);

    // prefetch next K block into registers (reload current on last iter)
    const int jn = (j + 32 <= qmax) ? j + 32 : j;
    const __bf16* Kn = Kb + (size_t)jn * HD;
    bk0 = load_frag(Kn, HD);
    bk1 = load_frag(Kn + 32, HD);
    bk2 = load_frag(Kn + 16 * HD, HD);
    bk3 = load_frag(Kn + 16 * HD + 32, HD);
    __builtin_prefetch(Kb + (size_t)(jn < qmax - 32 ? jn + 32 : jn) * HD + lane * 32, 0, 3);

    // V fragments for current block (independent of softmax below)
    v16bf bv0 = load_frag(Vb + j, SS);
    v16bf bv1 = load_frag(Vb + 16 * SS + j, SS);
    v16bf bv2 = load_frag(Vb + 32 * SS + j, SS);
    v16bf bv3 = load_frag(Vb + 48 * SS + j, SS);

#pragma unroll
    for (int i = 0; i < 8; ++i) {
      int q  = qbase + hlf * 8 + i;              // global query position
      int k0 = j + nl, k1 = k0 + 16;
      int i0 = min(max(q - k0, 0), 127);
      int i1 = min(max(q - k1, 0), 127);
      float x0 = (k0 <= q) ? (s0[i] * 0.125f + lut[i0]) : -1e9f;
      float x1 = (k1 <= q) ? (s1[i] * 0.125f + lut[i1]) : -1e9f;

      float mx = fmaxf(x0, x1);                  // row-max across 16-lane half
      mx = fmaxf(mx, __shfl_xor(mx, 1, 32));
      mx = fmaxf(mx, __shfl_xor(mx, 2, 32));
      mx = fmaxf(mx, __shfl_xor(mx, 4, 32));
      mx = fmaxf(mx, __shfl_xor(mx, 8, 32));
      float mn = fmaxf(m[i], mx);
      float corr = __expf(m[i] - mn);
      float p0 = __expf(x0 - mn);
      float p1 = __expf(x1 - mn);
      float rs = p0 + p1;
      rs += __shfl_xor(rs, 1, 32);
      rs += __shfl_xor(rs, 2, 32);
      rs += __shfl_xor(rs, 4, 32);
      rs += __shfl_xor(rs, 8, 32);
      l[i] = l[i] * corr + rs;
      m[i] = mn;
#pragma unroll
      for (int t = 0; t < 4; ++t) acc[t][i] *= corr;

      int row = hlf * 8 + i;                     // C-layout -> LDS row-major
      myp[row * 32 + nl]      = (__bf16)p0;
      myp[row * 32 + nl + 16] = (__bf16)p1;
    }

    // P(16x32) @ V(32x64): A from per-wave LDS (no barrier needed)
    v16bf ap = load_frag(myp, 32);
    acc[0] = wmma_bf16(ap, bv0, acc[0]);
    acc[1] = wmma_bf16(ap, bv1, acc[1]);
    acc[2] = wmma_bf16(ap, bv2, acc[2]);
    acc[3] = wmma_bf16(ap, bv3, acc[3]);
  }

#pragma unroll
  for (int i = 0; i < 8; ++i) {
    float rl = 1.0f / l[i];
    int s = qbase + hlf * 8 + i;
    size_t row = (size_t)b * SS + s;
#pragma unroll
    for (int t = 0; t < 4; ++t)
      ctx[row * DDIM + h * HD + t * 16 + nl] = (__bf16)(acc[t][i] * rl);
  }
}

// ---------------------------------------------------------------------------
extern "C" void kernel_launch(void* const* d_in, const int* in_sizes, int n_in,
                              void* d_out, int out_size, void* d_ws, size_t ws_size,
                              hipStream_t stream) {
  (void)in_sizes; (void)n_in; (void)out_size; (void)ws_size;
  const float* x   = (const float*)d_in[0];
  const float* Wq  = (const float*)d_in[1];
  const float* Wk  = (const float*)d_in[2];
  const float* Wv  = (const float*)d_in[3];
  const float* Wo  = (const float*)d_in[4];
  const float* rel = (const float*)d_in[5];

  char* ws = (char*)d_ws;
  __bf16* xbf = (__bf16*)(ws);                              // 8 MB
  __bf16* WqT = (__bf16*)(ws + (8ull << 20));               // 2 MB each
  __bf16* WkT = (__bf16*)(ws + (8ull << 20) + (2ull << 20));
  __bf16* WvT = (__bf16*)(ws + (8ull << 20) + (4ull << 20));
  __bf16* WoT = (__bf16*)(ws + (8ull << 20) + (6ull << 20));
  __bf16* Qbf = (__bf16*)(ws + (16ull << 20));              // 8 MB
  __bf16* Kbf = (__bf16*)(ws + (24ull << 20));              // 8 MB
  __bf16* Vt  = (__bf16*)(ws + (32ull << 20));              // 8 MB
  __bf16* ctx = (__bf16*)(ws + (40ull << 20));              // 8 MB

  cvt_bf16_kernel<<<(MROWS * DDIM) / 256, 256, 0, stream>>>(x, xbf, MROWS * DDIM);
  transpose_w_kernel<<<(DDIM * DDIM) / 256, 256, 0, stream>>>(Wq, WqT);
  transpose_w_kernel<<<(DDIM * DDIM) / 256, 256, 0, stream>>>(Wk, WkT);
  transpose_w_kernel<<<(DDIM * DDIM) / 256, 256, 0, stream>>>(Wv, WvT);
  transpose_w_kernel<<<(DDIM * DDIM) / 256, 256, 0, stream>>>(Wo, WoT);

  gemm_bf16_wmma_kernel<<<1024, 128, 0, stream>>>(xbf, WqT, Qbf, 0);  // Q + RoPE
  gemm_bf16_wmma_kernel<<<1024, 128, 0, stream>>>(xbf, WkT, Kbf, 0);  // K + RoPE
  gemm_bf16_wmma_kernel<<<1024, 128, 0, stream>>>(xbf, WvT, Vt, 1);   // V^T

  attn_wmma_kernel<<<dim3(BB * HH, SS / 64), 128, 0, stream>>>(
      Qbf, Kbf, Vt, rel, ctx);

  gemm_bf16_wmma_kernel<<<1024, 128, 0, stream>>>(ctx, WoT, d_out, 2);  // @Wo
}
